// DSCNN_RNN_Block_72129680769271
// MI455X (gfx1250) — matrix operand order repaired and checked by reading, not compiled, not run on hardware
//
#include <hip/hip_runtime.h>
#include <hip/hip_bf16.h>

// ---------------------------------------------------------------------------
// MI455X (gfx1250) DSCNN + bidirectional FastGRNN block.
// All matrix contractions via v_wmma_f32_16x16x32_bf16 (bf16 in, f32 accum).
// All GEMM B-operands are N-major (K contiguous) -> 32B vector loads, no
// divergence in inner loops. Activations use hardware transcendentals
// (v_exp_f32 / v_rcp_f32), branchless. Shapes hardcoded so idiv/imod fold.
// ---------------------------------------------------------------------------

typedef __bf16 bf16;
typedef __attribute__((ext_vector_type(16))) __bf16 v16bf;
typedef __attribute__((ext_vector_type(8)))  __bf16 v8bf;
typedef __attribute__((ext_vector_type(8)))  float  v8f;
typedef __attribute__((ext_vector_type(4)))  float  v4f;

#define B_    16
#define T_    1998
#define NBT   31968     // B_*T_  (divisible by 16 -> 1998 N-tiles)
#define FEAT  80
#define CH    256
#define RK    64
#define KIM   416       // im2col K (400 padded to multiple of 32)
#define NBF   662
#define NBB   664
#define LN    667       // 5+662 == 3+664
#define NN    10672     // B_*LN (divisible by 16 -> 667 N-tiles)

// Branchless fast activations: v_exp_f32 + v_rcp_f32, correct at +/-inf.
__device__ __forceinline__ float sigm(float x) {
  return __builtin_amdgcn_rcpf(1.0f + __expf(-x));
}
__device__ __forceinline__ float tanh_fast(float x) {
  return 1.0f - 2.0f * __builtin_amdgcn_rcpf(__expf(2.0f * x) + 1.0f);
}

__device__ __forceinline__ v8f wmma_bf16(v16bf a, v16bf b, v8f c) {
  return __builtin_amdgcn_wmma_f32_16x16x32_bf16(false, a, false, b, (short)0, c, false, false);
}

// A fragment (16x32 tile) from row-major float matrix, f32->bf16 on load.
// ISA 7.12.2 layout: lanes0-15 rows, K 0..7 & 16..23; lanes16-31 K 8..15 & 24..31.
// Requires lda % 4 == 0 (true for 64/256/128/416).
__device__ __forceinline__ v16bf load_a_f32(const float* __restrict__ A, int lda,
                                            int mbase, int kbase, int lane) {
  int row = mbase + (lane & 15);
  int kb  = kbase + ((lane >> 4) << 3);
  const v4f* p = (const v4f*)(A + (size_t)row * lda + kb);
  v4f f0 = p[0], f1 = p[1], f2 = p[4], f3 = p[5];
  v16bf a;
#pragma unroll
  for (int j = 0; j < 4; ++j) {
    a[j]      = (bf16)f0[j];
    a[4 + j]  = (bf16)f1[j];
    a[8 + j]  = (bf16)f2[j];
    a[12 + j] = (bf16)f3[j];
  }
  return a;
}

// B fragment (32x16 tile) from N-major bf16 (Bt[n*ldb + k]); K contiguous ->
// two 16-byte vector loads (K 8h..8h+7 & 16+8h..). Requires (ldb*2) % 32 == 0 (ldb in {64,256,416,128}).
__device__ __forceinline__ v16bf load_b_nt(const bf16* __restrict__ Bt, int ldb,
                                           int nbase, int kbase, int lane) {
  int col = nbase + (lane & 15);
  int kb  = kbase + ((lane >> 4) << 3);
  const v8bf* p = (const v8bf*)(Bt + (size_t)col * ldb + kb);
  v8bf b0 = p[0], b1 = p[2];
  v16bf b;
#pragma unroll
  for (int j = 0; j < 8; ++j) { b[j] = b0[j]; b[8 + j] = b1[j]; }
  return b;
}

// B fragment where B[k][n] = W[n][k], W row-major float (K contiguous).
__device__ __forceinline__ v16bf load_b_wt(const float* __restrict__ W, int ldw,
                                           int nbase, int kbase, int lane) {
  int col = nbase + (lane & 15);
  int kb  = kbase + ((lane >> 4) << 3);
  const v4f* p = (const v4f*)(W + (size_t)col * ldw + kb);
  v4f f0 = p[0], f1 = p[1], f2 = p[4], f3 = p[5];
  v16bf b;
#pragma unroll
  for (int j = 0; j < 4; ++j) {
    b[j]      = (bf16)f0[j];
    b[4 + j]  = (bf16)f1[j];
    b[8 + j]  = (bf16)f2[j];
    b[12 + j] = (bf16)f3[j];
  }
  return b;
}

// ---------------------------------------------------------------------------
// Prep kernels: zero-padded im2col (N-major bf16) and zero-padded W2 (f32).
// xim[n*KIM + kk] = feat[b, kk/5, t + kk%5 - 2]  (0 outside), n = b*T_ + t.
// ---------------------------------------------------------------------------
__global__ void __launch_bounds__(256)
im2col_k(const float* __restrict__ feat, bf16* __restrict__ xim) {
  size_t idx = (size_t)blockIdx.x * 256 + threadIdx.x;
  if (idx >= (size_t)NBT * KIM) return;
  int n  = (int)(idx / KIM);
  int kk = (int)(idx % KIM);
  int b = n / T_, t = n % T_;
  float v = 0.0f;
  if (kk < 400) {
    int cin = kk / 5, r = kk % 5;
    int tt = t + r - 2;
    if (tt >= 0 && tt < T_) v = feat[(b * FEAT + cin) * T_ + tt];
  }
  xim[idx] = (bf16)v;
}

__global__ void __launch_bounds__(256)
w2pad_k(const float* __restrict__ W2, float* __restrict__ W2p) {
  int idx = blockIdx.x * 256 + threadIdx.x;
  if (idx >= RK * KIM) return;
  int r = idx / KIM, kk = idx % KIM;
  W2p[idx] = (kk < 400) ? W2[r * 400 + kk] : 0.0f;
}

// ---------------------------------------------------------------------------
// Generic WMMA GEMM:  D[m,n] = sum_k A[m,k]*Bt[n,k]  (+bias[m])
// OL: 0 = bf16 out[n*ldo+m] (vector store); 1 = f32 out[m*ldo+n];
//     3 = f32 out[n*ldo+m] (vector store)
// grid.x = N/16, grid.y*4 waves = M/16. K % 32 == 0.
// ---------------------------------------------------------------------------
template <int OL, bool BIAS>
__global__ void __launch_bounds__(128)
wmma_gemm_k(const float* __restrict__ A, const bf16* __restrict__ Bt,
            void* __restrict__ Out, const float* __restrict__ bias,
            int K, int lda, int ldb, int ldo) {
  int lane  = threadIdx.x & 31;
  int wave  = threadIdx.x >> 5;
  int mbase = (blockIdx.y * 4 + wave) * 16;
  int nbase = blockIdx.x * 16;
  v8f acc = {};
  for (int kb = 0; kb < K; kb += 32) {
    v16bf af = load_a_f32(A, lda, mbase, kb, lane);
    v16bf bf = load_b_nt(Bt, ldb, nbase, kb, lane);
    acc = wmma_bf16(af, bf, acc);
  }
  int col  = nbase + (lane & 15);
  int rofs = (lane < 16) ? 0 : 8;
  int m0   = mbase + rofs;
  if (OL == 0) {
    v8bf o;
#pragma unroll
    for (int i = 0; i < 8; ++i) {
      float v = acc[i];
      if (BIAS) v += bias[m0 + i];
      o[i] = (bf16)v;
    }
    *(v8bf*)((bf16*)Out + (size_t)col * ldo + m0) = o;
  } else if (OL == 3) {
    v4f o0, o1;
#pragma unroll
    for (int i = 0; i < 4; ++i) {
      o0[i] = acc[i]     + (BIAS ? bias[m0 + i]     : 0.0f);
      o1[i] = acc[4 + i] + (BIAS ? bias[m0 + 4 + i] : 0.0f);
    }
    float* op = (float*)Out + (size_t)col * ldo + m0;
    *(v4f*)op = o0;
    *(v4f*)(op + 4) = o1;
  } else {  // OL == 1: f32, m-major (coalesced across lanes)
#pragma unroll
    for (int i = 0; i < 8; ++i) {
      float v = acc[i];
      if (BIAS) v += bias[m0 + i];
      ((float*)Out)[(size_t)(m0 + i) * ldo + col] = v;
    }
  }
}

// ---------------------------------------------------------------------------
// CNN1 stage 2 fused with tanh-gate:
//   pre[m,n] = sum_k W1[m,k] z1[n,k] + b[m]  (M=512)
//   xg[n,c]  = sigmoid(pre[c,n]) * tanh(pre[c+256,n])
// ---------------------------------------------------------------------------
__global__ void __launch_bounds__(128)
conv1_gate_k(const float* __restrict__ W1, const float* __restrict__ bias,
             const bf16* __restrict__ z1, bf16* __restrict__ xg) {
  int lane  = threadIdx.x & 31;
  int wave  = threadIdx.x >> 5;
  int mbase = (blockIdx.y * 4 + wave) * 16;   // sigmoid channel tile (0..255)
  int nbase = blockIdx.x * 16;
  v8f accS = {}, accT = {};
  for (int kb = 0; kb < RK; kb += 32) {
    v16bf aS = load_a_f32(W1, RK, mbase, kb, lane);
    v16bf aT = load_a_f32(W1, RK, mbase + 256, kb, lane);
    v16bf bf = load_b_nt(z1, RK, nbase, kb, lane);
    accS = wmma_bf16(aS, bf, accS);
    accT = wmma_bf16(aT, bf, accT);
  }
  int col  = nbase + (lane & 15);
  int rofs = (lane < 16) ? 0 : 8;
  int c0   = mbase + rofs;
  v8bf o;
#pragma unroll
  for (int i = 0; i < 8; ++i) {
    float s = sigm(accS[i] + bias[c0 + i]);
    float t = tanh_fast(accT[i] + bias[c0 + i + 256]);
    o[i] = (bf16)(s * t);
  }
  *(v8bf*)(xg + (size_t)col * CH + c0) = o;
}

// ---------------------------------------------------------------------------
// FastGRNN scan. One wave per block, 16 independent brick-sequences per wave.
// h (16x256 f32) in LDS; per step:
//   u1  = h  @ U1^T (16x64,  K=256) -> WMMA, staged in LDS as bf16
//   pre = u1 @ U2^T (16x256, K=64)  -> WMMA, fused gate update of h
// Writes last-step h (all bricks) + brick-0 / last-brick ::3 samples straight
// into xcat (512 x NN channel-major); COFF = 0 (fwd) / 256 (bwd).
// ---------------------------------------------------------------------------
template <int CTX, int NB, int DIR, int COFF>
__global__ void __launch_bounds__(32)
fastgrnn_scan_k(const float* __restrict__ xwu, const float* __restrict__ U1,
                const float* __restrict__ U2, const float* __restrict__ bg,
                const float* __restrict__ bu, const float* __restrict__ zeta_p,
                const float* __restrict__ nu_p, float* __restrict__ xcat) {
  __shared__ float hsm[16 * 256];
  __shared__ bf16  u1sm[16 * 64];
  __shared__ float bgs[256], bus[256];
  int lane = threadIdx.x;
  int s0   = blockIdx.x * 16;
  float zeta = sigm(zeta_p[0]);
  float nu   = sigm(nu_p[0]);
  for (int i = lane; i < 16 * 256; i += 32) hsm[i] = 0.0f;
  for (int i = lane; i < 256; i += 32) { bgs[i] = bg[i]; bus[i] = bu[i]; }
  int rofs = (lane < 16) ? 0 : 8;
  // Hoisted per-row bases (rows owned by this lane in the C/D layout).
  int xwbase[8];   // (b*T_ + j*3)*CH for xwu gather
  int ocbase[8];   // b*LN for xcat column base
  int jarr[8];
#pragma unroll
  for (int i = 0; i < 8; ++i) {
    int s = s0 + i + rofs;
    int b = s / NB, j = s % NB;
    xwbase[i] = (b * T_ + j * 3) * CH;
    ocbase[i] = b * LN;
    jarr[i]   = j;
  }
  __syncthreads();

  for (int tau = 0; tau < CTX; ++tau) {
    int tprime = DIR ? (CTX - 1 - tau) : tau;
    // ---- u1 = h @ U1^T --------------------------------------------------
#pragma unroll
    for (int nt = 0; nt < 4; ++nt) {
      v8f acc = {};
#pragma unroll
      for (int kb = 0; kb < 256; kb += 32) {
        int row = lane & 15;
        int kas = kb + ((lane >> 4) << 3);
        const v4f* hp = (const v4f*)(hsm + row * 256 + kas);
        v4f f0 = hp[0], f1 = hp[1], f2 = hp[4], f3 = hp[5];
        v16bf af;
#pragma unroll
        for (int j = 0; j < 4; ++j) {
          af[j]      = (bf16)f0[j];
          af[4 + j]  = (bf16)f1[j];
          af[8 + j]  = (bf16)f2[j];
          af[12 + j] = (bf16)f3[j];
        }
        v16bf bf = load_b_wt(U1, 256, nt * 16, kb, lane);  // B[k][r] = U1[r][k]
        acc = wmma_bf16(af, bf, acc);
      }
      int col = nt * 16 + (lane & 15);
#pragma unroll
      for (int i = 0; i < 8; ++i) u1sm[(i + rofs) * 64 + col] = (bf16)acc[i];
    }
    __syncthreads();

    // ---- pre = u1 @ U2^T, fused gate & h update -------------------------
    for (int nt = 0; nt < 16; ++nt) {
      v8f acc = {};
#pragma unroll
      for (int kb = 0; kb < 64; kb += 32) {
        int row = lane & 15;
        int kas = kb + ((lane >> 4) << 3);
        const v8bf* up = (const v8bf*)(u1sm + row * 64 + kas);
        v8bf b0 = up[0], b1 = up[2];
        v16bf af;
#pragma unroll
        for (int j = 0; j < 8; ++j) { af[j] = b0[j]; af[8 + j] = b1[j]; }
        v16bf bf = load_b_wt(U2, 64, nt * 16, kb, lane);   // B[r][h] = U2[h][r]
        acc = wmma_bf16(af, bf, acc);
      }
      int col = nt * 16 + (lane & 15);
      float bgc = bgs[col], buc = bus[col];
#pragma unroll
      for (int i = 0; i < 8; ++i) {
        int rs = i + rofs;
        float pre  = acc[i] + xwu[(size_t)(xwbase[i] + tprime * CH) + col];
        float z    = sigm(pre + bgc);
        float c    = tanh_fast(pre + buc);
        float hnew = z * hsm[rs * 256 + col] + (zeta * (1.0f - z) + nu) * c;
        hsm[rs * 256 + col] = hnew;
        size_t base = (size_t)(COFF + col) * NN + ocbase[i];
        if (tau == CTX - 1) xcat[base + (DIR ? jarr[i] : (jarr[i] + 5))] = hnew;
        if (DIR == 0) {
          if (jarr[i] == 0 && (tau % 3) == 0) xcat[base + tau / 3] = hnew;
        } else {
          if (jarr[i] == NB - 1 && tau < CTX - 1 && (tau % 3) == 0)
            xcat[base + (LN - 1 - tau / 3)] = hnew;
        }
      }
    }
    __syncthreads();
  }
}

// ---------------------------------------------------------------------------
// BatchNorm batch statistics (biased variance), one block per channel row.
// ---------------------------------------------------------------------------
__global__ void __launch_bounds__(256)
bn_stats_k(const float* __restrict__ x, float* __restrict__ mean,
           float* __restrict__ istd) {
  int c = blockIdx.x;
  const float* row = x + (size_t)c * NN;
  float s = 0.f, sq = 0.f;
  for (int i = threadIdx.x; i < NN; i += 256) { float v = row[i]; s += v; sq += v * v; }
  __shared__ float sh0[256], sh1[256];
  sh0[threadIdx.x] = s; sh1[threadIdx.x] = sq;
  __syncthreads();
  for (int off = 128; off > 0; off >>= 1) {
    if (threadIdx.x < off) { sh0[threadIdx.x] += sh0[threadIdx.x + off];
                             sh1[threadIdx.x] += sh1[threadIdx.x + off]; }
    __syncthreads();
  }
  if (threadIdx.x == 0) {
    float m = sh0[0] * (1.0f / NN);
    float v = sh1[0] * (1.0f / NN) - m * m;
    mean[c] = m;
    istd[c] = rsqrtf(v + 1e-5f);
  }
}

// tanh-gate (optionally pre-normalized by given stats). x: (2*ch, NN) -> g: (ch, NN)
__global__ void __launch_bounds__(256)
tanhgate_k(const float* __restrict__ x, const float* __restrict__ mean,
           const float* __restrict__ istd, float* __restrict__ g,
           int ch, int prenorm) {
  int idx = blockIdx.x * 256 + threadIdx.x;
  if (idx >= ch * NN) return;
  int c = idx / NN, n = idx % NN;
  float a = x[(size_t)c * NN + n];
  float b = x[(size_t)(c + ch) * NN + n];
  if (prenorm) { a = (a - mean[c]) * istd[c]; b = (b - mean[c + ch]) * istd[c + ch]; }
  g[idx] = sigm(a) * tanh_fast(b);
}

// Depthwise conv k=5 pad=2 along L, fused BN-normalize of input; output N-major
// bf16 d[n*ch + c] so the following rank GEMM gets contiguous K.
__global__ void __launch_bounds__(256)
dw_k(const float* __restrict__ g, const float* __restrict__ mean,
     const float* __restrict__ istd, const float* __restrict__ w,
     const float* __restrict__ bias, bf16* __restrict__ d, int ch) {
  int idx = blockIdx.x * 256 + threadIdx.x;
  if (idx >= ch * NN) return;
  int c = idx / NN, n = idx % NN;
  int b = n / LN, l = n % LN;
  float m = mean[c], is = istd[c];
  float acc = bias[c];
#pragma unroll
  for (int k = 0; k < 5; ++k) {
    int ll = l + k - 2;
    if (ll >= 0 && ll < LN)
      acc += w[c * 5 + k] * ((g[(size_t)c * NN + b * LN + ll] - m) * is);
  }
  d[(size_t)n * ch + c] = (bf16)acc;
}

// (256, NN) channel-major -> d_out (B, 256, LN)
__global__ void __launch_bounds__(256)
out_tr_k(const float* __restrict__ x, float* __restrict__ out) {
  int idx = blockIdx.x * 256 + threadIdx.x;
  if (idx >= 256 * NN) return;
  int c = idx / NN, n = idx % NN;
  int b = n / LN, l = n % LN;
  out[((size_t)b * 256 + c) * LN + l] = x[idx];
}

// ---------------------------------------------------------------------------
extern "C" void kernel_launch(void* const* d_in, const int* in_sizes, int n_in,
                              void* d_out, int out_size, void* d_ws, size_t ws_size,
                              hipStream_t stream) {
  (void)in_sizes; (void)n_in; (void)out_size; (void)ws_size;
  // JAX pytree (alphabetical dict) flattening order:
  const float* feat = (const float*)d_in[0];
  const float* c1W1 = (const float*)d_in[1];   // (512, 64)
  const float* c1W2 = (const float*)d_in[2];   // (64, 400)
  const float* c1b  = (const float*)d_in[3];   // (512,)
  struct Dscnn { const float *dw_b, *dw_w, *pw_W1, *pw_W2, *pw_b; } blk[4];
  for (int i = 0; i < 4; ++i) {
    int b0 = 4 + 5 * i;  // keys: dw_b, dw_w, pw_W1, pw_W2, pw_b
    blk[i] = { (const float*)d_in[b0], (const float*)d_in[b0 + 1],
               (const float*)d_in[b0 + 2], (const float*)d_in[b0 + 3],
               (const float*)d_in[b0 + 4] };
  }
  struct Rnn { const float *U1, *U2, *W1, *W2, *bg, *bu, *nu, *zeta; } rb, rf;
  { int b0 = 24; rb = { (const float*)d_in[b0], (const float*)d_in[b0+1],
                        (const float*)d_in[b0+2], (const float*)d_in[b0+3],
                        (const float*)d_in[b0+4], (const float*)d_in[b0+5],
                        (const float*)d_in[b0+6], (const float*)d_in[b0+7] }; }
  { int b0 = 32; rf = { (const float*)d_in[b0], (const float*)d_in[b0+1],
                        (const float*)d_in[b0+2], (const float*)d_in[b0+3],
                        (const float*)d_in[b0+4], (const float*)d_in[b0+5],
                        (const float*)d_in[b0+6], (const float*)d_in[b0+7] }; }

  // ---- workspace carve-up -------------------------------------------------
  char* base = (char*)d_ws;
  size_t off = 0;
  auto carve = [&](size_t bytes) -> char* {
    char* p = base + off; off += (bytes + 255) & ~(size_t)255; return p;
  };
  bf16*  xim   = (bf16*) carve((size_t)NBT * KIM * 2);
  float* W2p   = (float*)carve((size_t)RK * KIM * 4);
  bf16*  z1    = (bf16*) carve((size_t)NBT * RK * 2);
  bf16*  xg    = (bf16*) carve((size_t)NBT * CH * 2);
  bf16*  tmpr  = (bf16*) carve((size_t)NBT * RK * 2);
  float* xwu_f = (float*)carve((size_t)NBT * CH * 4);
  float* xwu_b = (float*)carve((size_t)NBT * CH * 4);
  float* xcat  = (float*)carve((size_t)512 * NN * 4);
  float* meanb = (float*)carve(512 * 4);
  float* istdb = (float*)carve(512 * 4);
  float* gbuf  = (float*)carve((size_t)256 * NN * 4);
  bf16*  dbuf  = (bf16*) carve((size_t)256 * NN * 2);
  bf16*  t2buf = (bf16*) carve((size_t)RK * NN * 2);
  float* pbufA = (float*)carve((size_t)256 * NN * 4);
  float* pbufB = (float*)carve((size_t)256 * NN * 4);

  // ---- CNN1: low-rank conv k5 pad2 + tanh-gate ----------------------------
  {
    size_t tot = (size_t)NBT * KIM;
    im2col_k<<<dim3((unsigned)((tot + 255) / 256)), 256, 0, stream>>>(feat, xim);
    w2pad_k<<<dim3((RK * KIM + 255) / 256), 256, 0, stream>>>(c1W2, W2p);
  }
  wmma_gemm_k<0, false><<<dim3(NBT / 16, 1), 128, 0, stream>>>(
      W2p, xim, z1, nullptr, KIM, KIM, KIM, RK);
  conv1_gate_k<<<dim3(NBT / 16, 4), 128, 0, stream>>>(c1W1, c1b, z1, xg);

  // ---- xw precompute (per unique time step), both directions --------------
  // tmp[n,r] = sum_d xg[n,d]*W1[r,d]; xwu[n,h] = sum_r tmp[n,r]*W2[h,r]
  wmma_gemm_k<0, false><<<dim3(NBT / 16, 1), 128, 0, stream>>>(
      rf.W1, xg, tmpr, nullptr, CH, CH, CH, RK);
  wmma_gemm_k<3, false><<<dim3(NBT / 16, 4), 128, 0, stream>>>(
      rf.W2, tmpr, xwu_f, nullptr, RK, RK, RK, CH);
  wmma_gemm_k<0, false><<<dim3(NBT / 16, 1), 128, 0, stream>>>(
      rb.W1, xg, tmpr, nullptr, CH, CH, CH, RK);
  wmma_gemm_k<3, false><<<dim3(NBT / 16, 4), 128, 0, stream>>>(
      rb.W2, tmpr, xwu_b, nullptr, RK, RK, RK, CH);

  // ---- FastGRNN scans, writing post-RNN concat directly -------------------
  fastgrnn_scan_k<15, NBF, 0, 0><<<dim3((B_ * NBF) / 16), 32, 0, stream>>>(
      xwu_f, rf.U1, rf.U2, rf.bg, rf.bu, rf.zeta, rf.nu, xcat);
  fastgrnn_scan_k<9, NBB, 1, 256><<<dim3((B_ * NBB) / 16), 32, 0, stream>>>(
      xwu_b, rb.U1, rb.U2, rb.bg, rb.bu, rb.zeta, rb.nu, xcat);

  // ---- bnorm_rnn + 4 DSCNN blocks -----------------------------------------
  const float* cur = xcat;
  int curC = 512;
  for (int i = 0; i < 4; ++i) {
    int ch = curC / 2;
    if (i == 0) bn_stats_k<<<dim3(512), 256, 0, stream>>>(xcat, meanb, istdb);
    int grid_e = (ch * NN + 255) / 256;
    tanhgate_k<<<dim3(grid_e), 256, 0, stream>>>(cur, meanb, istdb, gbuf, ch,
                                                 (i == 0) ? 1 : 0);
    bn_stats_k<<<dim3(ch), 256, 0, stream>>>(gbuf, meanb, istdb);
    dw_k<<<dim3(grid_e), 256, 0, stream>>>(gbuf, meanb, istdb, blk[i].dw_w,
                                           blk[i].dw_b, dbuf, ch);
    // low-rank pointwise: t2[n,r] = sum_c W2[r,c] d[n,c]; out = W1 @ t2 + b
    wmma_gemm_k<0, false><<<dim3(NN / 16, 1), 128, 0, stream>>>(
        blk[i].pw_W2, dbuf, t2buf, nullptr, ch, ch, ch, RK);
    float* outp = (i & 1) ? pbufB : pbufA;
    wmma_gemm_k<1, true><<<dim3(NN / 16, 4), 128, 0, stream>>>(
        blk[i].pw_W1, t2buf, outp, blk[i].pw_b, RK, RK, RK, NN);
    cur = outp;
    curC = 256;
  }

  // ---- transpose (256, NN) -> (B, 256, LN) --------------------------------
  out_tr_k<<<dim3((256 * NN + 255) / 256), 256, 0, stream>>>(cur, (float*)d_out);
}
